// TiGer_model_3607772529226
// MI455X (gfx1250) — matrix operand order repaired and checked by reading, not compiled, not run on hardware
//
#include <hip/hip_runtime.h>
#include <hip/hip_bf16.h>

// ---------------------------------------------------------------------------
// Problem constants (match reference)
// ---------------------------------------------------------------------------
#define NN   120000
#define FF   256
#define HH   256
#define LL   64
#define EE   200000
#define EMP  1000000
#define P0   60000
#define P1   80000
#define P2   100000
#define PROX_W 0.3f

typedef __attribute__((ext_vector_type(16))) __bf16 v16bf;
typedef __attribute__((ext_vector_type(8)))  float  v8f;
typedef __attribute__((ext_vector_type(4)))  unsigned int v4u;
typedef __attribute__((ext_vector_type(8)))  int v8i;
typedef __attribute__((ext_vector_type(4)))  int v4i;

__device__ __forceinline__ unsigned short f2bf(float f) {
    unsigned u = __builtin_bit_cast(unsigned, f);
    unsigned r = (u + 0x7FFFu + ((u >> 16) & 1u)) >> 16;   // RNE
    return (unsigned short)r;
}

// ---------------------------------------------------------------------------
// TDM: issue TENSOR_LOAD_TO_LDS for a 2D tile (bf16 elements).
// D# built per CDNA5 ISA §8.3/8.4:
//   group0: count=1 | lds_addr | global_addr(57b) | type=2
//   group1: wg_mask=0, data_size=1(2B), tensor_dim0/1, tile_dim0/1,
//           tensor_dim0_stride; dims in data_size units
//   groups 2/3: zero (2D tensor)
// Rows beyond tensor_dim1 read as zero (hardware OOB padding).
// ---------------------------------------------------------------------------
__device__ __forceinline__ void tdm_load_tile_bf16(
    unsigned lds_byte_addr, const void* gaddr,
    unsigned tensor_d0, unsigned tensor_d1,
    unsigned tile_d0, unsigned tile_d1,
    unsigned long long d0_stride)
{
    unsigned long long ga = (unsigned long long)gaddr;
    v4u g0;
    g0[0] = 1u;                                       // count=1, user D#, no gather
    g0[1] = lds_byte_addr;                            // lds_addr [63:32]
    g0[2] = (unsigned)(ga & 0xFFFFFFFFu);             // global_addr lo
    g0[3] = (unsigned)((ga >> 32) & 0x01FFFFFFu)      // global_addr hi (57b total)
            | (2u << 30);                             // type=2 ("image")
    v8i g1;
    g1[0] = (int)(1u << 16);                          // wg_mask=0 | data_size=1 (2B)
    g1[1] = (int)((tensor_d0 & 0xFFFFu) << 16);       // abar_addr=0 | tdim0[15:0]
    g1[2] = (int)((tensor_d0 >> 16) | ((tensor_d1 & 0xFFFFu) << 16));
    g1[3] = (int)((tensor_d1 >> 16) | (tile_d0 << 16));          // tdim1 hi | tile_dim0
    g1[4] = (int)(tile_d1 & 0xFFFFu);                            // tile_dim1 | tile_dim2=0
    g1[5] = (int)(d0_stride & 0xFFFFFFFFu);                      // dim0_stride lo
    g1[6] = (int)((d0_stride >> 32) & 0xFFFFu);                  // stride hi | dim1_stride lo=0
    g1[7] = 0;
    v4i z = {0, 0, 0, 0};
#if defined(__clang_major__) && __clang_major__ >= 23
    v8i z8 = {0, 0, 0, 0, 0, 0, 0, 0};
    __builtin_amdgcn_tensor_load_to_lds(g0, g1, z, z, z8, 0);
#else
    __builtin_amdgcn_tensor_load_to_lds(g0, g1, z, z, 0);
#endif
}

// ---------------------------------------------------------------------------
// Elementwise helpers
// ---------------------------------------------------------------------------
__global__ void k_fill(float* p, float v, long n) {
    long i = (long)blockIdx.x * blockDim.x + threadIdx.x;
    if (i < n) p[i] = v;
}

__global__ void k_cast_bf16(const float* __restrict__ s, unsigned short* __restrict__ d, long n) {
    long i = (long)blockIdx.x * blockDim.x + threadIdx.x;
    if (i < n) d[i] = f2bf(s[i]);
}

__global__ void k_deg_atomic(const int* __restrict__ dst, float* deg, int e) {
    long i = (long)blockIdx.x * blockDim.x + threadIdx.x;
    if (i < e) atomicAdd(&deg[dst[i]], 1.0f);
}

__global__ void k_rsqrt_inplace(float* deg, int n) {
    long i = (long)blockIdx.x * blockDim.x + threadIdx.x;
    if (i < n) deg[i] = rsqrtf(fmaxf(deg[i], 1.0f));
}

// out[i,f] = xw[i,f] * dinv[i]^2   (self-loop term of GCN aggregation)
__global__ void k_selfterm(const float* __restrict__ xw, const float* __restrict__ dinv,
                           float* __restrict__ out, long n) {
    long i = (long)blockIdx.x * blockDim.x + threadIdx.x;
    if (i >= n) return;
    int row = (int)(i >> 8);
    float dv = dinv[row];
    out[i] = xw[i] * dv * dv;
}

// edge scatter: out[dst] += xw[src] * dinv[src]*dinv[dst]  (4 feats / thread)
__global__ void k_scatter(const float* __restrict__ xw, const float* __restrict__ dinv,
                          const int* __restrict__ src, const int* __restrict__ dst,
                          float* __restrict__ out, int e) {
    long idx = (long)blockIdx.x * blockDim.x + threadIdx.x;
    long total = (long)e * 64;
    if (idx >= total) return;
    int ei = (int)(idx >> 6);
    int f4 = (int)(idx & 63) * 4;
    int s = src[ei], d = dst[ei];
    float nrm = dinv[s] * dinv[d];
    const float4 v = *(const float4*)(xw + (long)s * HH + f4);
    float* o = out + (long)d * HH + f4;
    atomicAdd(o + 0, v.x * nrm);
    atomicAdd(o + 1, v.y * nrm);
    atomicAdd(o + 2, v.z * nrm);
    atomicAdd(o + 3, v.w * nrm);
}

// out = tanh(in + b[f]); also write bf16 copy
__global__ void k_bias_tanh_cast(const float* __restrict__ in, const float* __restrict__ b,
                                 float* __restrict__ out, unsigned short* __restrict__ obf, long n) {
    long i = (long)blockIdx.x * blockDim.x + threadIdx.x;
    if (i >= n) return;
    float v = tanhf(in[i] + b[(int)(i & 255)]);
    out[i] = v;
    obf[i] = f2bf(v);
}

__global__ void k_sym_bf16(const float* __restrict__ W, unsigned short* __restrict__ S) {
    int i = blockIdx.x * blockDim.x + threadIdx.x;
    if (i >= HH * HH) return;
    int r = i >> 8, c = i & 255;
    S[i] = f2bf(0.5f * (W[r * HH + c] + W[c * HH + r]));
}

// ---------------------------------------------------------------------------
// bf16 WMMA GEMM: C[M,Nc] = act(A[M,K] @ B[K,Nc] + bias)
// Block = 256 threads = 8 waves. Block tile 128(M) x 128(N), BK = 32.
// Wave (wm in 0..3, wn in 0..1) owns a 32x64 patch = 2x4 WMMA accumulators
// (8 v_wmma per wave per K-step; A frags reused 4x, B frags reused 2x).
//
// A tile (128x32 bf16, row-major in LDS) is loaded by the Tensor Data Mover
// (TENSOR_LOAD_TO_LDS, TENSORcnt), issued by wave 0 each K-step; hardware
// zero-pads rows past M via tensor_dim1 = M - bM.
// B tile is software-swizzled: sB[(col*2 + khalf)*16 + e] = B[k0+khalf*16+e][bN+col]
// so each lane's 16 B elements are one contiguous 32B run (b128 frag reads).
// ACT: 0 = none, 1 = tanh
// ---------------------------------------------------------------------------
template <int ACT>
__global__ __launch_bounds__(256)
void gemm_bf16(const unsigned short* __restrict__ A, const unsigned short* __restrict__ B,
               const float* __restrict__ bias, float* __restrict__ C,
               int M, int K, int Nc) {
    __shared__ unsigned short sA[128 * 32];      // 8 KB (TDM destination)
    __shared__ unsigned short sB[128 * 2 * 16];  // 8 KB (swizzled)

    const int tid  = threadIdx.x;
    const int lane = tid & 31;
    const int wave = tid >> 5;
    const int wm   = wave & 3;   // 0..3 -> M offset 32*wm
    const int wn   = wave >> 2;  // 0..1 -> N offset 64*wn
    const int bM   = blockIdx.y * 128;
    const int bN   = blockIdx.x * 128;
    const int l16   = lane & 15;
    const int khalf = lane >> 4;

    // LDS byte address of sA (flat LDS-aperture address truncates to offset)
    const unsigned ldsA = (unsigned)(unsigned long long)(const void*)sA;

    // B staging coords: unit = 4 rows x 4 cols per thread
    const int bcol  = (tid & 31) * 4;    // 0..124
    const int brow  = (tid >> 5) * 4;    // 0..28
    const int bkh   = brow >> 4;         // 0 or 1
    const int beb   = brow & 15;         // element base (multiple of 4)

    v8f acc[2][4] = {};

    for (int k0 = 0; k0 < K; k0 += 32) {
        // --- stage A tile via Tensor Data Mover (one wave issues the DMA) ---
        if (tid < 32) {
            tdm_load_tile_bf16(ldsA, (const void*)(A + (long)bM * K + k0),
                               (unsigned)K, (unsigned)(M - bM),
                               32u, 128u, (unsigned long long)K);
            __builtin_amdgcn_s_wait_tensorcnt(0);
        }
        // --- stage B tile (32x128 bf16) swizzled: 4 rows x 4 cols per thread ---
        {
            uint2 g0 = *(const uint2*)(B + (long)(k0 + brow)     * Nc + bN + bcol);
            uint2 g1 = *(const uint2*)(B + (long)(k0 + brow + 1) * Nc + bN + bcol);
            uint2 g2 = *(const uint2*)(B + (long)(k0 + brow + 2) * Nc + bN + bcol);
            uint2 g3 = *(const uint2*)(B + (long)(k0 + brow + 3) * Nc + bN + bcol);
            // column c+cc gets ushorts {rows r..r+3} as two dwords
#pragma unroll
            for (int cc = 0; cc < 4; ++cc) {
                unsigned w0 = (cc < 2) ? g0.x : g0.y;
                unsigned w1 = (cc < 2) ? g1.x : g1.y;
                unsigned w2 = (cc < 2) ? g2.x : g2.y;
                unsigned w3 = (cc < 2) ? g3.x : g3.y;
                unsigned d0, d1;
                if (cc & 1) {
                    d0 = (w0 >> 16) | (w1 & 0xFFFF0000u);
                    d1 = (w2 >> 16) | (w3 & 0xFFFF0000u);
                } else {
                    d0 = (w0 & 0xFFFFu) | (w1 << 16);
                    d1 = (w2 & 0xFFFFu) | (w3 << 16);
                }
                uint2 dv = make_uint2(d0, d1);
                *(uint2*)(sB + ((bcol + cc) * 2 + bkh) * 16 + beb) = dv;
            }
        }
        // prefetch next B tile rows (global_prefetch_b8)
        if (k0 + 32 < K)
            __builtin_prefetch(B + (long)(k0 + 32 + brow) * Nc + bN + bcol, 0, 3);
        __syncthreads();

        // --- fragments ---
        v16bf afrag[2], bfrag[4];
#pragma unroll
        for (int mi = 0; mi < 2; ++mi) {
            int mrow = wm * 32 + mi * 16 + l16;
            // elements 0..7:  k = khalf*16 + e        (contiguous 16B)
            // elements 8..15: k = khalf*16 + e        (contiguous 16B)
#pragma unroll
            for (int e = 0; e < 8; ++e)
                afrag[mi][e] = __builtin_bit_cast(__bf16, sA[mrow * 32 + khalf * 16 + e]);
#pragma unroll
            for (int e = 8; e < 16; ++e)
                afrag[mi][e] = __builtin_bit_cast(__bf16, sA[mrow * 32 + khalf * 16 + e]);
        }
#pragma unroll
        for (int ni = 0; ni < 4; ++ni) {
            int ncol = wn * 64 + ni * 16 + l16;
            const unsigned short* bp = sB + (ncol * 2 + khalf) * 16;  // 16 contiguous ushorts
#pragma unroll
            for (int e = 0; e < 16; ++e)
                bfrag[ni][e] = __builtin_bit_cast(__bf16, bp[e]);
        }
#pragma unroll
        for (int mi = 0; mi < 2; ++mi)
#pragma unroll
            for (int ni = 0; ni < 4; ++ni)
                acc[mi][ni] = __builtin_amdgcn_wmma_f32_16x16x32_bf16(
                    false, afrag[mi], false, bfrag[ni], (short)0, acc[mi][ni], false, false);
        __syncthreads();
    }

    // --- epilogue: D layout -> VGPR r, lanes0-15: M=r ; lanes16-31: M=8+r ---
    const int mhigh = khalf * 8;
#pragma unroll
    for (int mi = 0; mi < 2; ++mi) {
#pragma unroll
        for (int ni = 0; ni < 4; ++ni) {
            int col = bN + wn * 64 + ni * 16 + l16;
            float bv = bias ? bias[col] : 0.0f;
#pragma unroll
            for (int r = 0; r < 8; ++r) {
                int row = bM + wm * 32 + mi * 16 + mhigh + r;
                if (row < M) {
                    float v = acc[mi][ni][r] + bv;
                    if (ACT == 1) v = tanhf(v);
                    C[(long)row * Nc + col] = v;
                }
            }
        }
    }
}

// ---------------------------------------------------------------------------
// Attention / edge kernels (wave32 reductions)
// ---------------------------------------------------------------------------
__device__ __forceinline__ float wave_sum(float s) {
    for (int o = 16; o; o >>= 1) s += __shfl_xor(s, o, 32);
    return s;
}

// dots[r] = (q[r] . kmat[r]) / sqrt(H)
__global__ void k_dotq(const float* __restrict__ q, const float* __restrict__ kmat,
                       float* __restrict__ dots, int nwin) {
    int row  = blockIdx.x * 8 + (threadIdx.x >> 5);
    int lane = threadIdx.x & 31;
    if (row >= nwin) return;
    const float* qa = q + (long)row * HH;
    const float* ka = kmat + (long)row * HH;
    float s = 0.0f;
    for (int f = lane; f < HH; f += 32) s += qa[f] * ka[f];
    s = wave_sum(s);
    if (lane == 0) dots[row] = s * 0.0625f;   // 1/sqrt(256)
}

// softmax over S snapshots; dots strided by `stride`
__global__ void k_softmax_w(const float* __restrict__ dots, float* __restrict__ wts,
                            int nwin, int S, int stride) {
    int r = blockIdx.x * blockDim.x + threadIdx.x;
    if (r >= nwin) return;
    float m = -1e30f;
    for (int s = 0; s < S; ++s) m = fmaxf(m, dots[s * stride + r]);
    float e[4], sum = 0.0f;
    for (int s = 0; s < S; ++s) { e[s] = expf(dots[s * stride + r] - m); sum += e[s]; }
    float inv = 1.0f / sum;
    for (int s = 0; s < S; ++s) wts[r * 4 + s] = e[s] * inv;
}

// attn[i] (s==0 ? = : +=) wts[r,s] * v[i]
__global__ void k_vaccum(const float* __restrict__ v, const float* __restrict__ wts,
                         float* __restrict__ attn, long n, int s) {
    long i = (long)blockIdx.x * blockDim.x + threadIdx.x;
    if (i >= n) return;
    float w = wts[(i >> 8) * 4 + s];
    float x = w * v[i];
    if (s == 0) attn[i] = x; else attn[i] += x;
}

// cat[row, 0:256) = bf(emb), cat[row, 256:512) = bf(attn)
__global__ void k_build_cat(const float* __restrict__ emb, const float* __restrict__ attn,
                            unsigned short* __restrict__ cat, long n) {
    long i = (long)blockIdx.x * blockDim.x + threadIdx.x;
    if (i >= n) return;
    long row = i >> 8; int f = (int)(i & 255);
    cat[row * 512 + f]        = f2bf(emb[i]);
    cat[row * 512 + 256 + f]  = f2bf(attn[i]);
}

// sim = dot(embsym[es], emb2[ed]) + sum(bias);  mlscore = sigmoid(sim)
__global__ void k_edge_sim(const float* __restrict__ embsym, const float* __restrict__ emb2,
                           const float* __restrict__ bias, const int* __restrict__ es,
                           const int* __restrict__ ed, float* __restrict__ mlscore, int e) {
    int ei   = blockIdx.x * 8 + (threadIdx.x >> 5);
    int lane = threadIdx.x & 31;
    if (ei >= e) return;
    const float* a = embsym + (long)es[ei] * HH;
    const float* b = emb2  + (long)ed[ei] * HH;
    float s = 0.0f;
    for (int f = lane; f < HH; f += 32) s += a[f] * b[f] + bias[f];
    s = wave_sum(s);
    if (lane == 0) mlscore[ei] = 1.0f / (1.0f + expf(-s));
}

// feat[e, 0:256) = bf(mean), feat[e, 256:512) = bf(max)
__global__ void k_build_feat(const float* __restrict__ src, const int* __restrict__ es,
                             const int* __restrict__ ed, unsigned short* __restrict__ feat, int e) {
    long i = (long)blockIdx.x * blockDim.x + threadIdx.x;
    if (i >= (long)e * HH) return;
    int ei = (int)(i >> 8), f = (int)(i & 255);
    float a = src[(long)es[ei] * HH + f];
    float b = src[(long)ed[ei] * HH + f];
    feat[(long)ei * 512 + f]       = f2bf((a + b) * 0.5f);
    feat[(long)ei * 512 + 256 + f] = f2bf(fmaxf(a, b));
}

// out[e] = tanh(dot(hid[e], w2) + b2)
__global__ void k_dot_w2(const float* __restrict__ hid, const float* __restrict__ w2,
                         const float* __restrict__ b2, float* __restrict__ out, int e) {
    int ei   = blockIdx.x * 8 + (threadIdx.x >> 5);
    int lane = threadIdx.x & 31;
    if (ei >= e) return;
    const float* h = hid + (long)ei * HH;
    float s = 0.0f;
    for (int f = lane; f < HH; f += 32) s += h[f] * w2[f];
    s = wave_sum(s);
    if (lane == 0) out[ei] = tanhf(s + b2[0]);
}

__global__ void k_final(const float* __restrict__ a_ml, const float* __restrict__ a_ms,
                        const float* __restrict__ mlscore, const float* __restrict__ ms_ts,
                        const float* __restrict__ px_ts, const int* __restrict__ index,
                        float* __restrict__ out, int e) {
    int ei = blockIdx.x * blockDim.x + threadIdx.x;
    if (ei >= e) return;
    float a0 = a_ml[ei], a1 = a_ms[ei], a2 = PROX_W;
    float m = fmaxf(a0, fmaxf(a1, a2));
    float e0 = expf(a0 - m), e1 = expf(a1 - m), e2 = expf(a2 - m);
    float inv = 1.0f / (e0 + e1 + e2);
    int idx = index[ei];
    float v = (mlscore[ei] * e0 + ms_ts[idx] * e1 + px_ts[idx] * e2) * inv;
    out[ei] = fminf(fmaxf(v, 0.0f), 1.0f);
}

// ---------------------------------------------------------------------------
// Host orchestration
// ---------------------------------------------------------------------------
static inline int nblk(long n, int t) { return (int)((n + t - 1) / t); }

extern "C" void kernel_launch(void* const* d_in, const int* in_sizes, int n_in,
                              void* d_out, int out_size, void* d_ws, size_t ws_size,
                              hipStream_t stream) {
    (void)in_sizes; (void)n_in; (void)out_size; (void)ws_size;

    const float* x        = (const float*)d_in[0];
    const int*   mp       = (const int*)  d_in[1];
    const int*   edges    = (const int*)  d_in[2];
    const int*   index    = (const int*)  d_in[3];
    const float* prev0    = (const float*)d_in[4];
    const float* prev1    = (const float*)d_in[5];
    const float* prev2    = (const float*)d_in[6];
    const float* gc1_W    = (const float*)d_in[7];
    const float* gc1_b    = (const float*)d_in[8];
    const float* gc2_W    = (const float*)d_in[9];
    const float* gc2_b    = (const float*)d_in[10];
    const float* lin_W    = (const float*)d_in[11];
    const float* lin_b    = (const float*)d_in[12];
    const float* weight_lin = (const float*)d_in[13];
    const float* bias_lin = (const float*)d_in[14];
    const float* w_q      = (const float*)d_in[15];
    const float* w_k      = (const float*)d_in[16];
    const float* w_v      = (const float*)d_in[17];
    const float* ml_w1    = (const float*)d_in[18];
    const float* ml_b1    = (const float*)d_in[19];
    const float* ml_w2    = (const float*)d_in[20];
    const float* ml_b2    = (const float*)d_in[21];
    const float* ms_w1    = (const float*)d_in[22];
    const float* ms_b1    = (const float*)d_in[23];
    const float* ms_w2    = (const float*)d_in[24];
    const float* ms_b2    = (const float*)d_in[25];
    const float* msdr_w   = (const float*)d_in[26];
    const float* msdr_b   = (const float*)d_in[27];
    const float* logits   = (const float*)d_in[28];
    const float* ms_ts    = (const float*)d_in[29];
    const float* px_ts    = (const float*)d_in[30];
    float* out = (float*)d_out;

    const int* src_mp = mp;
    const int* dst_mp = mp + EMP;
    const int* es     = edges;
    const int* ed     = edges + EE;

    // ---- workspace carve-out (deterministic fixed offsets) ----
    char* ws = (char*)d_ws;
    size_t off = 0;
    auto alloc = [&](size_t bytes) -> void* {
        void* p = ws + off;
        off += (bytes + 255) & ~(size_t)255;
        return p;
    };
    unsigned short* xbf     = (unsigned short*)alloc((size_t)NN * FF * 2);
    unsigned short* p0bf    = (unsigned short*)alloc((size_t)P0 * HH * 2);
    unsigned short* p1bf    = (unsigned short*)alloc((size_t)P1 * HH * 2);
    unsigned short* p2bf    = (unsigned short*)alloc((size_t)P2 * HH * 2);
    unsigned short* logbf   = (unsigned short*)alloc((size_t)NN * LL * 2);
    unsigned short* gc1Wbf  = (unsigned short*)alloc((size_t)FF * HH * 2);
    unsigned short* gc2Wbf  = (unsigned short*)alloc((size_t)HH * HH * 2);
    unsigned short* wqbf    = (unsigned short*)alloc((size_t)HH * HH * 2);
    unsigned short* wkbf    = (unsigned short*)alloc((size_t)HH * HH * 2);
    unsigned short* wvbf    = (unsigned short*)alloc((size_t)HH * HH * 2);
    unsigned short* linWbf  = (unsigned short*)alloc((size_t)2 * HH * HH * 2);
    unsigned short* mlw1bf  = (unsigned short*)alloc((size_t)2 * HH * HH * 2);
    unsigned short* msw1bf  = (unsigned short*)alloc((size_t)2 * HH * HH * 2);
    unsigned short* msdrWbf = (unsigned short*)alloc((size_t)LL * HH * 2);
    unsigned short* symbf   = (unsigned short*)alloc((size_t)HH * HH * 2);
    float* dinv    = (float*)alloc((size_t)NN * 4);
    float* xw      = (float*)alloc((size_t)NN * HH * 4);
    float* agg     = (float*)alloc((size_t)NN * HH * 4);
    float* hbuf    = (float*)alloc((size_t)NN * HH * 4);
    unsigned short* hbf   = (unsigned short*)alloc((size_t)NN * HH * 2);
    float* embbuf  = (float*)alloc((size_t)NN * HH * 4);
    unsigned short* embbf = (unsigned short*)alloc((size_t)NN * HH * 2);
    float* qbuf    = (float*)alloc((size_t)NN * HH * 4);
    float* ktmp    = (float*)alloc((size_t)P0 * HH * 4);       // max window = 60000 rows
    float* dots    = (float*)alloc((size_t)4 * P0 * 4);
    float* wts     = (float*)alloc((size_t)P0 * 4 * 4);
    float* attnbuf = (float*)alloc((size_t)NN * HH * 4);
    unsigned short* catbf  = (unsigned short*)alloc((size_t)NN * 2 * HH * 2);
    float* emb2    = (float*)alloc((size_t)NN * HH * 4);
    unsigned short* emb2bf = (unsigned short*)alloc((size_t)NN * HH * 2);
    float* embsym  = (float*)alloc((size_t)NN * HH * 4);
    float* rbuf    = (float*)alloc((size_t)NN * HH * 4);
    unsigned short* featbf = (unsigned short*)alloc((size_t)EE * 2 * HH * 2);
    float* hid     = (float*)alloc((size_t)EE * HH * 4);
    float* mlscore = (float*)alloc((size_t)EE * 4);
    float* a_ml    = (float*)alloc((size_t)EE * 4);
    float* a_ms    = (float*)alloc((size_t)EE * 4);

    const int T = 256;
    // ---- bf16 casts ----
    k_cast_bf16<<<nblk((long)NN * FF, T), T, 0, stream>>>(x, xbf, (long)NN * FF);
    k_cast_bf16<<<nblk((long)P0 * HH, T), T, 0, stream>>>(prev0, p0bf, (long)P0 * HH);
    k_cast_bf16<<<nblk((long)P1 * HH, T), T, 0, stream>>>(prev1, p1bf, (long)P1 * HH);
    k_cast_bf16<<<nblk((long)P2 * HH, T), T, 0, stream>>>(prev2, p2bf, (long)P2 * HH);
    k_cast_bf16<<<nblk((long)NN * LL, T), T, 0, stream>>>(logits, logbf, (long)NN * LL);
    k_cast_bf16<<<nblk(FF * HH, T), T, 0, stream>>>(gc1_W, gc1Wbf, FF * HH);
    k_cast_bf16<<<nblk(HH * HH, T), T, 0, stream>>>(gc2_W, gc2Wbf, HH * HH);
    k_cast_bf16<<<nblk(HH * HH, T), T, 0, stream>>>(w_q, wqbf, HH * HH);
    k_cast_bf16<<<nblk(HH * HH, T), T, 0, stream>>>(w_k, wkbf, HH * HH);
    k_cast_bf16<<<nblk(HH * HH, T), T, 0, stream>>>(w_v, wvbf, HH * HH);
    k_cast_bf16<<<nblk(2 * HH * HH, T), T, 0, stream>>>(lin_W, linWbf, 2 * HH * HH);
    k_cast_bf16<<<nblk(2 * HH * HH, T), T, 0, stream>>>(ml_w1, mlw1bf, 2 * HH * HH);
    k_cast_bf16<<<nblk(2 * HH * HH, T), T, 0, stream>>>(ms_w1, msw1bf, 2 * HH * HH);
    k_cast_bf16<<<nblk(LL * HH, T), T, 0, stream>>>(msdr_w, msdrWbf, LL * HH);
    k_sym_bf16<<<nblk(HH * HH, T), T, 0, stream>>>(weight_lin, symbf);

    // ---- GCN degree normalization: deg = in-degree + 1 (self loop) ----
    k_fill<<<nblk(NN, T), T, 0, stream>>>(dinv, 1.0f, NN);
    k_deg_atomic<<<nblk(EMP, T), T, 0, stream>>>(dst_mp, dinv, EMP);
    k_rsqrt_inplace<<<nblk(NN, T), T, 0, stream>>>(dinv, NN);

    dim3 g256(HH / 128, (NN + 127) / 128);
    const long NH = (long)NN * HH;

    // ---- GCN layer 1: h = tanh(norm-agg(x @ W1) + b1) ----
    gemm_bf16<0><<<g256, T, 0, stream>>>(xbf, gc1Wbf, nullptr, xw, NN, FF, HH);
    k_selfterm<<<nblk(NH, T), T, 0, stream>>>(xw, dinv, agg, NH);
    k_scatter<<<nblk((long)EMP * 64, T), T, 0, stream>>>(xw, dinv, src_mp, dst_mp, agg, EMP);
    k_bias_tanh_cast<<<nblk(NH, T), T, 0, stream>>>(agg, gc1_b, hbuf, hbf, NH);

    // ---- GCN layer 2: emb = tanh(norm-agg(h @ W2) + b2) ----
    gemm_bf16<0><<<g256, T, 0, stream>>>(hbf, gc2Wbf, nullptr, xw, NN, HH, HH);
    k_selfterm<<<nblk(NH, T), T, 0, stream>>>(xw, dinv, agg, NH);
    k_scatter<<<nblk((long)EMP * 64, T), T, 0, stream>>>(xw, dinv, src_mp, dst_mp, agg, EMP);
    k_bias_tanh_cast<<<nblk(NH, T), T, 0, stream>>>(agg, gc2_b, embbuf, embbf, NH);

    // ---- temporal attention: q = emb @ w_q ----
    gemm_bf16<0><<<g256, T, 0, stream>>>(embbf, wqbf, nullptr, qbuf, NN, HH, HH);

    struct Win { int start, end, S; const unsigned short* snaps[4]; };
    Win wins[3] = {
        {0,      P0, 4, {p0bf, p1bf, p2bf, embbf}},
        {P0,     P1, 3, {p1bf, p2bf, embbf, nullptr}},
        {P1,     P2, 2, {p2bf, embbf, nullptr, nullptr}},
    };
    for (int wi = 0; wi < 3; ++wi) {
        const Win& W = wins[wi];
        int nwin = W.end - W.start;
        dim3 gw(HH / 128, (nwin + 127) / 128);
        for (int s = 0; s < W.S; ++s) {
            gemm_bf16<0><<<gw, T, 0, stream>>>(W.snaps[s] + (long)W.start * HH, wkbf,
                                               nullptr, ktmp, nwin, HH, HH);
            k_dotq<<<nblk(nwin, 8), T, 0, stream>>>(qbuf + (long)W.start * HH, ktmp,
                                                    dots + (long)s * P0, nwin);
        }
        k_softmax_w<<<nblk(nwin, T), T, 0, stream>>>(dots, wts, nwin, W.S, P0);
        for (int s = 0; s < W.S; ++s) {
            gemm_bf16<0><<<gw, T, 0, stream>>>(W.snaps[s] + (long)W.start * HH, wvbf,
                                               nullptr, ktmp, nwin, HH, HH);
            k_vaccum<<<nblk((long)nwin * HH, T), T, 0, stream>>>(ktmp, wts,
                attnbuf + (long)W.start * HH, (long)nwin * HH, s);
        }
    }
    // tail [100000, 120000): attn = emb @ w_v
    {
        int nt = NN - P2;
        dim3 gt(HH / 128, (nt + 127) / 128);
        gemm_bf16<0><<<gt, T, 0, stream>>>(embbf + (long)P2 * HH, wvbf, nullptr,
                                           attnbuf + (long)P2 * HH, nt, HH, HH);
    }

    // ---- emb2 = tanh([emb | attn] @ linear_W + b) ----
    k_build_cat<<<nblk(NH, T), T, 0, stream>>>(embbuf, attnbuf, catbf, NH);
    gemm_bf16<1><<<g256, T, 0, stream>>>(catbf, linWbf, lin_b, emb2, NN, 2 * HH, HH);
    k_cast_bf16<<<nblk(NH, T), T, 0, stream>>>(emb2, emb2bf, NH);

    // ---- bilinear edge scores: embsym = emb2 @ sym (per node), then per-edge dot ----
    gemm_bf16<0><<<g256, T, 0, stream>>>(emb2bf, symbf, nullptr, embsym, NN, HH, HH);
    k_edge_sim<<<nblk(EE, 8), T, 0, stream>>>(embsym, emb2, bias_lin, es, ed, mlscore, EE);

    // ---- r = tanh(logits @ msdr_w + b) per node ----
    gemm_bf16<1><<<g256, T, 0, stream>>>(logbf, msdrWbf, msdr_b, rbuf, NN, LL, HH);

    // ---- ensemble ML head ----
    k_build_feat<<<nblk((long)EE * HH, T), T, 0, stream>>>(emb2, es, ed, featbf, EE);
    {
        dim3 ge(HH / 128, (EE + 127) / 128);
        gemm_bf16<1><<<ge, T, 0, stream>>>(featbf, mlw1bf, ml_b1, hid, EE, 2 * HH, HH);
    }
    k_dot_w2<<<nblk(EE, 8), T, 0, stream>>>(hid, ml_w2, ml_b2, a_ml, EE);

    // ---- ensemble MS head ----
    k_build_feat<<<nblk((long)EE * HH, T), T, 0, stream>>>(rbuf, es, ed, featbf, EE);
    {
        dim3 ge(HH / 128, (EE + 127) / 128);
        gemm_bf16<1><<<ge, T, 0, stream>>>(featbf, msw1bf, ms_b1, hid, EE, 2 * HH, HH);
    }
    k_dot_w2<<<nblk(EE, 8), T, 0, stream>>>(hid, ms_w2, ms_b2, a_ms, EE);

    // ---- final softmax ensemble + clip ----
    k_final<<<nblk(EE, T), T, 0, stream>>>(a_ml, a_ms, mlscore, ms_ts, px_ts, index, out, EE);
}
